// MyGAT_23278722744834
// MI455X (gfx1250) — matrix operand order repaired and checked by reading, not compiled, not run on hardware
//
#include <hip/hip_runtime.h>
#include <hip/hip_bf16.h>
#include <math.h>

#define NN      50000
#define NE      800000
#define ETOT    (NE + NN)
#define HC      256          // heads*hid = 256 in both layers
#define NEG_SLOPE 0.2f

typedef __attribute__((ext_vector_type(16))) __bf16 v16bf;
typedef __attribute__((ext_vector_type(8)))  __bf16 v8bf;
typedef __attribute__((ext_vector_type(8)))  float  v8f;

__device__ __forceinline__ __bf16 f2bf(float f) {
    unsigned u = __float_as_uint(f);
    unsigned r = u + 0x7FFFu + ((u >> 16) & 1u);   // round-to-nearest-even
    unsigned short h = (unsigned short)(r >> 16);
    __bf16 b;
    __builtin_memcpy(&b, &h, sizeof(h));
    return b;
}

__global__ void k_cast_bf16(const float* __restrict__ in, __bf16* __restrict__ out, int n) {
    int i = blockIdx.x * blockDim.x + threadIdx.x;
    if (i < n) out[i] = f2bf(in[i]);
}

// Wt[n][k] = bf16(W[k][n]);  W row-major [K][Ncols]
__global__ void k_cast_transpose(const float* __restrict__ W, __bf16* __restrict__ Wt,
                                 int K, int Ncols) {
    int i = blockIdx.x * blockDim.x + threadIdx.x;
    if (i >= K * Ncols) return;
    int k = i / Ncols, n2 = i - k * Ncols;
    Wt[(size_t)n2 * K + k] = f2bf(W[i]);
}

__global__ void k_fill(float* __restrict__ p, float v, int n) {
    int i = blockIdx.x * blockDim.x + threadIdx.x;
    if (i < n) p[i] = v;
}

// C[M x Ncols] = A[M x K](bf16 rowmajor) * B, with Bt[Ncols x K] = B^T (bf16), + optional bias.
// One wave computes a 16x64 strip: one shared A fragment, 4 B fragments, 4 WMMA accumulators.
// Waves are linearized over (M/16) * (Ncols/64) tiles; block = 128 threads (4 waves).
__global__ void __launch_bounds__(128)
k_gemm_bf16(const __bf16* __restrict__ A, const __bf16* __restrict__ Bt,
            const float* __restrict__ bias, float* __restrict__ C,
            int M, int K, int Ncols) {
    const int lane = threadIdx.x & 31;
    const int wid  = blockIdx.x * (blockDim.x >> 5) + (threadIdx.x >> 5);
    const int n64  = Ncols >> 6;                       // # of 64-wide col groups
    const int mtile = wid / n64;
    const int ng    = wid - mtile * n64;
    if (mtile * 16 >= M) return;                       // wave-uniform guard
    const int mbase = mtile * 16;
    const int nbase = ng * 64;

    // A 16x32: row=lane&15, lanes 0-15 hold K 0..15; lanes 16-31 hold K 16..31 (matches B)
    const int arow = mbase + (lane & 15);
    const int kA0  = (lane & 16) ? 16 : 0;
    // B 32x16: lane = column, lanes 0-15 K=0..15, lanes 16-31 K=16..31 (k-contiguous in Bt row)
    const int ncol0 = nbase + (lane & 15);
    const int kB0   = (lane & 16) ? 16 : 0;
    const __bf16* Ap = A  + (size_t)arow  * K + kA0;
    const __bf16* Bp = Bt + (size_t)ncol0 * K + kB0;

    v8f acc[4] = {};
    for (int k = 0; k < K; k += 32) {
        __builtin_prefetch(Ap + k + 64, 0, 1);
        v8bf alo = *(const v8bf*)(Ap + k);
        v8bf ahi = *(const v8bf*)(Ap + k + 8);
        v16bf a;
        #pragma unroll
        for (int j = 0; j < 8; ++j) { a[j] = alo[j]; a[j + 8] = ahi[j]; }
        #pragma unroll
        for (int t = 0; t < 4; ++t) {
            v16bf b = *(const v16bf*)(Bp + (size_t)(t * 16) * K + k);
            acc[t] = __builtin_amdgcn_wmma_f32_16x16x32_bf16(false, a, false, b,
                                                             (short)0, acc[t], false, false);
        }
    }
    // C/D f32 layout: VGPR r, lane L -> M = r + 8*(L>=16), N = L&15
    const int crow = mbase + ((lane & 16) ? 8 : 0);
    const int cc   = lane & 15;
    #pragma unroll
    for (int t = 0; t < 4; ++t) {
        const int ncol = nbase + t * 16 + cc;
        const float bv = bias ? bias[ncol] : 0.0f;
        #pragma unroll
        for (int r = 0; r < 8; ++r)
            C[(size_t)(crow + r) * Ncols + ncol] = acc[t][r] + bv;
    }
}

// Wave per node: lane covers 8 contiguous channels (fully coalesced 1KB row read),
// shfl_xor reduction over C/8-lane segments (4 lanes for H=8, 32 for H=1).
__global__ void k_attn_scores(const float* __restrict__ h, const float* __restrict__ a_src,
                              const float* __restrict__ a_dst, float* __restrict__ s,
                              float* __restrict__ d, int H, int C) {
    int wid = blockIdx.x * (blockDim.x >> 5) + (threadIdx.x >> 5);
    if (wid >= NN) return;
    int lane  = threadIdx.x & 31;
    int cbase = lane * 8;                  // channel in [0,256); a_src is [H][C] so flat index = cbase
    int hh    = cbase / C;
    const float* hp = h + (size_t)wid * HC + cbase;
    float4 h0 = *(const float4*)(hp);
    float4 h1 = *(const float4*)(hp + 4);
    float4 s0 = *(const float4*)(a_src + cbase);
    float4 s1 = *(const float4*)(a_src + cbase + 4);
    float4 d0 = *(const float4*)(a_dst + cbase);
    float4 d1 = *(const float4*)(a_dst + cbase + 4);
    float ss = h0.x*s0.x + h0.y*s0.y + h0.z*s0.z + h0.w*s0.w
             + h1.x*s1.x + h1.y*s1.y + h1.z*s1.z + h1.w*s1.w;
    float dd = h0.x*d0.x + h0.y*d0.y + h0.z*d0.z + h0.w*d0.w
             + h1.x*d1.x + h1.y*d1.y + h1.z*d1.z + h1.w*d1.w;
    int width = C >> 3;                    // lanes per head segment: 4 (H=8) or 32 (H=1)
    for (int off = width >> 1; off > 0; off >>= 1) {
        ss += __shfl_xor(ss, off, 32);
        dd += __shfl_xor(dd, off, 32);
    }
    if ((lane & (width - 1)) == 0) {
        s[wid * H + hh] = ss;
        d[wid * H + hh] = dd;
    }
}

__device__ __forceinline__ void edge_nodes(const int* __restrict__ ei, int e, int& src, int& dst) {
    if (e < NE) { src = ei[e]; dst = ei[NE + e]; }
    else        { src = dst = e - NE; }               // appended self-loops
}

// float atomic max via int/uint ordering trick (valid with -inf init)
__device__ __forceinline__ void atomicMaxFloat(float* addr, float val) {
    int iv = __float_as_int(val);
    if (iv >= 0) atomicMax((int*)addr, iv);
    else         atomicMin((unsigned int*)addr, (unsigned int)iv);
}

__global__ void k_edge_logits(const int* __restrict__ ei, const float* __restrict__ s,
                              const float* __restrict__ d, float* __restrict__ ev,
                              float* __restrict__ emax, int H) {
    int i = blockIdx.x * blockDim.x + threadIdx.x;
    if (i >= ETOT * H) return;
    int e = i / H, hh = i - e * H;
    int src, dst; edge_nodes(ei, e, src, dst);
    float v = s[src * H + hh] + d[dst * H + hh];
    v = (v > 0.f) ? v : NEG_SLOPE * v;                // leaky relu
    ev[i] = v;
    atomicMaxFloat(&emax[dst * H + hh], v);
}

__global__ void k_edge_exp(const int* __restrict__ ei, float* __restrict__ ev,
                           const float* __restrict__ emax, float* __restrict__ denom, int H) {
    int i = blockIdx.x * blockDim.x + threadIdx.x;
    if (i >= ETOT * H) return;
    int e = i / H, hh = i - e * H;
    int src, dst; edge_nodes(ei, e, src, dst);
    float v = __expf(ev[i] - emax[dst * H + hh]);
    ev[i] = v;
    atomicAdd(&denom[dst * H + hh], v);
}

// grid = ETOT blocks, 256 threads: thread = one channel of one edge (coalesced gather + atomic scatter)
__global__ void k_edge_scatter(const int* __restrict__ ei, const float* __restrict__ h,
                               const float* __restrict__ ev, const float* __restrict__ denom,
                               float* __restrict__ agg, int H, int C) {
    int e = blockIdx.x;
    int c = threadIdx.x;
    int src, dst; edge_nodes(ei, e, src, dst);
    int hh = c / C;
    float alpha = ev[e * H + hh] / denom[dst * H + hh];
    atomicAdd(&agg[(size_t)dst * HC + c], h[(size_t)src * HC + c] * alpha);
}

__global__ void k_bias_elu_cast(const float* __restrict__ agg, const float* __restrict__ bias,
                                __bf16* __restrict__ out, int n) {
    int i = blockIdx.x * blockDim.x + threadIdx.x;
    if (i >= n) return;
    int c = i & (HC - 1);
    float v = agg[i] + bias[c];
    v = (v > 0.f) ? v : (__expf(v) - 1.f);            // elu
    out[i] = f2bf(v);
}

extern "C" void kernel_launch(void* const* d_in, const int* in_sizes, int n_in,
                              void* d_out, int out_size, void* d_ws, size_t ws_size,
                              hipStream_t stream) {
    const float* x      = (const float*)d_in[0];
    const int*   ei     = (const int*)  d_in[1];
    const float* W1     = (const float*)d_in[2];
    const float* a_src1 = (const float*)d_in[3];
    const float* a_dst1 = (const float*)d_in[4];
    const float* b1     = (const float*)d_in[5];
    const float* W2     = (const float*)d_in[6];
    const float* a_src2 = (const float*)d_in[7];
    const float* a_dst2 = (const float*)d_in[8];
    const float* b2     = (const float*)d_in[9];
    const float* Wlin   = (const float*)d_in[10];
    const float* blin   = (const float*)d_in[11];
    float* out = (float*)d_out;

    char* ws = (char*)d_ws;
    size_t off = 0;
    auto alloc = [&](size_t bytes) -> void* {
        void* p = ws + off;
        off += (bytes + 255) & ~(size_t)255;
        return p;
    };
    __bf16* B_xh  = (__bf16*)alloc((size_t)NN * HC * 2);  // xh, later h1b/h2b
    __bf16* W1t   = (__bf16*)alloc(256 * 256 * 2);
    __bf16* W2t   = (__bf16*)alloc(256 * 256 * 2);
    __bf16* Wlt   = (__bf16*)alloc(256 * 64 * 2);
    float*  B_h   = (float*) alloc((size_t)NN * HC * 4);  // h1, later h2
    float*  B_agg = (float*) alloc((size_t)NN * HC * 4);
    float*  B_e   = (float*) alloc((size_t)ETOT * 8 * 4);
    float*  B_s   = (float*) alloc((size_t)NN * 8 * 4);
    float*  B_d   = (float*) alloc((size_t)NN * 8 * 4);
    float*  B_max = (float*) alloc((size_t)NN * 8 * 4);
    float*  B_den = (float*) alloc((size_t)NN * 8 * 4);
    (void)ws_size; (void)in_sizes; (void)n_in; (void)out_size;

    const int TB = 256;
    auto nb = [](long long n) { return (unsigned)((n + 255) / 256); };
    // GEMM grid: waves = (M/16)*(Ncols/64), 4 waves per 128-thread block
    auto gemm_grid = [](int M, int Ncols) {
        long long waves = (long long)((M + 15) / 16) * (Ncols >> 6);
        return dim3((unsigned)((waves + 3) / 4));
    };
    const unsigned attn_blocks = (NN + 7) / 8;            // 8 waves per 256-thread block
    const float NINF = -__builtin_inff();

    // ---- prep: bf16 casts / weight transposes ----
    k_cast_bf16<<<nb((long long)NN * HC), TB, 0, stream>>>(x, B_xh, NN * HC);
    k_cast_transpose<<<nb(256 * 256), TB, 0, stream>>>(W1,   W1t, 256, 256);
    k_cast_transpose<<<nb(256 * 256), TB, 0, stream>>>(W2,   W2t, 256, 256);
    k_cast_transpose<<<nb(256 * 64),  TB, 0, stream>>>(Wlin, Wlt, 256, 64);

    // ---- layer 1 (H=8, C=32) ----
    k_gemm_bf16<<<gemm_grid(NN, 256), 128, 0, stream>>>(B_xh, W1t, nullptr, B_h, NN, 256, 256);
    k_attn_scores<<<attn_blocks, TB, 0, stream>>>(B_h, a_src1, a_dst1, B_s, B_d, 8, 32);
    k_fill<<<nb((long long)NN * 8), TB, 0, stream>>>(B_max, NINF, NN * 8);
    k_fill<<<nb((long long)NN * 8), TB, 0, stream>>>(B_den, 0.f, NN * 8);
    k_fill<<<nb((long long)NN * HC), TB, 0, stream>>>(B_agg, 0.f, NN * HC);
    k_edge_logits<<<nb((long long)ETOT * 8), TB, 0, stream>>>(ei, B_s, B_d, B_e, B_max, 8);
    k_edge_exp<<<nb((long long)ETOT * 8), TB, 0, stream>>>(ei, B_e, B_max, B_den, 8);
    k_edge_scatter<<<ETOT, 256, 0, stream>>>(ei, B_h, B_e, B_den, B_agg, 8, 32);
    k_bias_elu_cast<<<nb((long long)NN * HC), TB, 0, stream>>>(B_agg, b1, B_xh, NN * HC);

    // ---- layer 2 (H=1, C=256) ----
    k_gemm_bf16<<<gemm_grid(NN, 256), 128, 0, stream>>>(B_xh, W2t, nullptr, B_h, NN, 256, 256);
    k_attn_scores<<<attn_blocks, TB, 0, stream>>>(B_h, a_src2, a_dst2, B_s, B_d, 1, 256);
    k_fill<<<nb(NN), TB, 0, stream>>>(B_max, NINF, NN);
    k_fill<<<nb(NN), TB, 0, stream>>>(B_den, 0.f, NN);
    k_fill<<<nb((long long)NN * HC), TB, 0, stream>>>(B_agg, 0.f, NN * HC);
    k_edge_logits<<<nb(ETOT), TB, 0, stream>>>(ei, B_s, B_d, B_e, B_max, 1);
    k_edge_exp<<<nb(ETOT), TB, 0, stream>>>(ei, B_e, B_max, B_den, 1);
    k_edge_scatter<<<ETOT, 256, 0, stream>>>(ei, B_h, B_e, B_den, B_agg, 1, 256);
    k_bias_elu_cast<<<nb((long long)NN * HC), TB, 0, stream>>>(B_agg, b2, B_xh, NN * HC);

    // ---- final linear: [50000,256] x [256,64] + blin -> d_out ----
    k_gemm_bf16<<<gemm_grid(NN, 64), 128, 0, stream>>>(B_xh, Wlt, blin, out, NN, 256, 64);
}
